// RNN_51857435132099
// MI455X (gfx1250) — matrix operand order repaired and checked by reading, not compiled, not run on hardware
//
#include <hip/hip_runtime.h>
#include <hip/hip_bf16.h>
#include <math.h>

// Model dims
#define B_SZ     512
#define SC_LEN   48
#define ST_LEN   16
#define E_DIM    512
#define H_DIM    256
#define N_CHARS_ 128
#define N_TAGS_  64
#define LANG_DIM_ 64

typedef __bf16 bf16;
typedef __attribute__((ext_vector_type(16))) __bf16 v16bf;
typedef __attribute__((ext_vector_type(8)))  __bf16 v8bf;
typedef __attribute__((ext_vector_type(4)))  __bf16 v4bf;
typedef __attribute__((ext_vector_type(8)))  float  v8f;

#define BM 128
#define BN 64
#define BK 32

__device__ __forceinline__ v8f zero_v8f() {
  v8f z = {0.f,0.f,0.f,0.f,0.f,0.f,0.f,0.f};
  return z;
}

__device__ __forceinline__ v4bf cvt4(const float4 f) {
  v4bf h;
  h[0] = (bf16)f.x; h[1] = (bf16)f.y; h[2] = (bf16)f.z; h[3] = (bf16)f.w;
  return h;
}

// C[M,N] = alpha * A[M,K] x B + bias[N]
//   TRANSB=true : B is [N,K] row-major (typical weight layout, X @ W^T)
//   TRANSB=false: B is [K,N] row-major
// Batched via blockIdx.z with element strides sA/sB/sC.
// bf16 WMMA 16x16x32, f32 accumulate. Block tile 128x64, 8 waves of 32x32.
// Regular tiles (full MN, K%32==0): double-buffered LDS + register prefetch,
// one barrier per K-chunk, b128 global loads overlapped with WMMA.
// Edge tiles (small attention matmuls) take a guarded scalar path.
template<bool TRANSB>
__global__ __launch_bounds__(256) void gemm_bf16_wmma(
    const float* __restrict__ A, const float* __restrict__ Bm,
    const float* __restrict__ bias, float* __restrict__ C,
    int M, int N, int K, int lda, int ldb, int ldc,
    long long sA, long long sB, long long sC, float alpha)
{
  __shared__ __align__(16) bf16 As[2][BM * BK];
  __shared__ __align__(16) bf16 Bs[2][BN * BK];   // transposed: Bs[.][n*BK + k]

  const int tileN = blockIdx.x * BN;
  const int tileM = blockIdx.y * BM;
  A  += (long long)blockIdx.z * sA;
  Bm += (long long)blockIdx.z * sB;
  C  += (long long)blockIdx.z * sC;

  const int tid  = threadIdx.x;
  const int lane = tid & 31;
  const int wave = tid >> 5;
  const int wm   = (wave >> 1) << 5;   // 4 waves along M
  const int wn   = (wave & 1) << 5;    // 2 waves along N
  const int half = lane >> 4;          // lane half (0/1)
  const int lm   = lane & 15;

  v8f acc[2][2];
  acc[0][0] = zero_v8f(); acc[0][1] = zero_v8f();
  acc[1][0] = zero_v8f(); acc[1][1] = zero_v8f();

  const int  nk      = (K + BK - 1) / BK;
  const bool regular = (tileM + BM <= M) && (tileN + BN <= N) && ((K & (BK - 1)) == 0);

  if (regular) {
    float4 ra[4], rb[2];
    auto loadRegs = [&](int k0) {
#pragma unroll
      for (int l = 0; l < 4; ++l) {              // A: 1024 quads / 256 thr
        int q = tid + l * 256;
        int r = q >> 3, c4 = (q & 7) << 2;       // 8 quads per 32-wide row
        ra[l] = *(const float4*)(A + (long long)(tileM + r) * lda + k0 + c4);
      }
      if (TRANSB) {
#pragma unroll
        for (int l = 0; l < 2; ++l) {            // B: 512 quads / 256 thr
          int q = tid + l * 256;
          int n = q >> 3, c4 = (q & 7) << 2;
          rb[l] = *(const float4*)(Bm + (long long)(tileN + n) * ldb + k0 + c4);
        }
      } else {
#pragma unroll
        for (int l = 0; l < 2; ++l) {            // B[K,N]: vector load along n
          int q = tid + l * 256;
          int k = q >> 4, n4 = (q & 15) << 2;    // 16 quads per k-row
          rb[l] = *(const float4*)(Bm + (long long)(k0 + k) * ldb + tileN + n4);
        }
      }
    };
    auto storeRegs = [&](int buf) {
      bf16* as = &As[buf][0];
      bf16* bs = &Bs[buf][0];
#pragma unroll
      for (int l = 0; l < 4; ++l) {
        int q = tid + l * 256;
        int r = q >> 3, c4 = (q & 7) << 2;
        *(v4bf*)(&as[r * BK + c4]) = cvt4(ra[l]);
      }
      if (TRANSB) {
#pragma unroll
        for (int l = 0; l < 2; ++l) {
          int q = tid + l * 256;
          int n = q >> 3, c4 = (q & 7) << 2;
          *(v4bf*)(&bs[n * BK + c4]) = cvt4(rb[l]);
        }
      } else {
#pragma unroll
        for (int l = 0; l < 2; ++l) {
          int q = tid + l * 256;
          int k = q >> 4, n4 = (q & 15) << 2;
          bs[(n4 + 0) * BK + k] = (bf16)rb[l].x;
          bs[(n4 + 1) * BK + k] = (bf16)rb[l].y;
          bs[(n4 + 2) * BK + k] = (bf16)rb[l].z;
          bs[(n4 + 3) * BK + k] = (bf16)rb[l].w;
        }
      }
    };

    loadRegs(0);
    for (int kc = 0; kc < nk; ++kc) {
      const int buf = kc & 1;
      storeRegs(buf);                 // stores chunk kc (waits outstanding loads)
      __syncthreads();                // single barrier per chunk (double buffer)
      if (kc + 1 < nk) loadRegs((kc + 1) * BK);   // prefetch overlaps WMMA below

      v16bf af[2], bfg[2];
#pragma unroll
      for (int i = 0; i < 2; ++i) {
        const bf16* arow = &As[buf][(wm + i * 16 + lm) * BK];
        v8bf lo = *(const v8bf*)(arow + half * 8);
        v8bf hi = *(const v8bf*)(arow + 16 + half * 8);
        af[i] = __builtin_shufflevector(lo, hi, 0,1,2,3,4,5,6,7,8,9,10,11,12,13,14,15);
      }
#pragma unroll
      for (int j = 0; j < 2; ++j) {
        const bf16* brow = &Bs[buf][(wn + j * 16 + lm) * BK];
        v8bf lo = *(const v8bf*)(brow + half * 8);
        v8bf hi = *(const v8bf*)(brow + 16 + half * 8);
        bfg[j] = __builtin_shufflevector(lo, hi, 0,1,2,3,4,5,6,7,8,9,10,11,12,13,14,15);
      }
#pragma unroll
      for (int i = 0; i < 2; ++i)
#pragma unroll
        for (int j = 0; j < 2; ++j)
          acc[i][j] = __builtin_amdgcn_wmma_f32_16x16x32_bf16(
              false, af[i], false, bfg[j], (short)0, acc[i][j], false, false);
    }
  } else {
    // ---- guarded scalar path (edge tiles only) ----
    for (int kc = 0; kc < nk; ++kc) {
      const int k0 = kc * BK;
      for (int i = tid; i < BM * BK; i += 256) {
        int r = i >> 5, c = i & 31;
        int gr = tileM + r, gc = k0 + c;
        float v = 0.f;
        if (gr < M && gc < K) v = A[(long long)gr * lda + gc];
        As[0][i] = (bf16)v;
      }
      for (int i = tid; i < BN * BK; i += 256) {
        int n = i >> 5, c = i & 31;
        int gn = tileN + n, gc = k0 + c;
        float v = 0.f;
        if (gn < N && gc < K)
          v = TRANSB ? Bm[(long long)gn * ldb + gc]
                     : Bm[(long long)gc * ldb + gn];
        Bs[0][i] = (bf16)v;
      }
      __syncthreads();

      v16bf af[2], bfg[2];
#pragma unroll
      for (int i = 0; i < 2; ++i) {
        const bf16* arow = &As[0][(wm + i * 16 + lm) * BK];
        v8bf lo = *(const v8bf*)(arow + half * 8);
        v8bf hi = *(const v8bf*)(arow + 16 + half * 8);
        af[i] = __builtin_shufflevector(lo, hi, 0,1,2,3,4,5,6,7,8,9,10,11,12,13,14,15);
      }
#pragma unroll
      for (int j = 0; j < 2; ++j) {
        const bf16* brow = &Bs[0][(wn + j * 16 + lm) * BK];
        v8bf lo = *(const v8bf*)(brow + half * 8);
        v8bf hi = *(const v8bf*)(brow + 16 + half * 8);
        bfg[j] = __builtin_shufflevector(lo, hi, 0,1,2,3,4,5,6,7,8,9,10,11,12,13,14,15);
      }
#pragma unroll
      for (int i = 0; i < 2; ++i)
#pragma unroll
        for (int j = 0; j < 2; ++j)
          acc[i][j] = __builtin_amdgcn_wmma_f32_16x16x32_bf16(
              false, af[i], false, bfg[j], (short)0, acc[i][j], false, false);
      __syncthreads();
    }
  }

  // C layout: lane half h, col lm; VGPR r => row h*8 + r
#pragma unroll
  for (int i = 0; i < 2; ++i) {
#pragma unroll
    for (int j = 0; j < 2; ++j) {
      int ncol = tileN + wn + j * 16 + lm;
      if (ncol >= N) continue;
      float bv = bias ? bias[ncol] : 0.f;
#pragma unroll
      for (int r = 0; r < 8; ++r) {
        int mrow = tileM + wm + i * 16 + half * 8 + r;
        if (mrow < M)
          C[(long long)mrow * ldc + ncol] = alpha * acc[i][j][r] + bv;
      }
    }
  }
}

// emb[b] = lang_embeds[ li[ li[b] ] ]  (faithful double index)
__global__ void build_emb_k(const int* __restrict__ li, const float* __restrict__ le,
                            float* __restrict__ emb) {
  int idx = blockIdx.x * blockDim.x + threadIdx.x;
  if (idx >= B_SZ * LANG_DIM_) return;
  int b = idx / LANG_DIM_, d = idx % LANG_DIM_;
  int l2 = li[li[b]];
  emb[idx] = le[l2 * LANG_DIM_ + d];
}

// out[s][b][0:F] = seq[b][s][:]; out[s][b][F:F+64] = emb[b][:]   (time-major)
__global__ void concat_tm_k(const float* __restrict__ seq, const float* __restrict__ emb,
                            float* __restrict__ out, int S, int F) {
  int I = F + LANG_DIM_;
  long long total = (long long)S * B_SZ * I;
  long long idx = (long long)blockIdx.x * blockDim.x + threadIdx.x;
  if (idx >= total) return;
  int i = (int)(idx % I);
  long long sb = idx / I;
  int b = (int)(sb % B_SZ);
  int s = (int)(sb / B_SZ);
  out[idx] = (i < F) ? seq[((long long)b * S + s) * F + i]
                     : emb[(long long)b * LANG_DIM_ + (i - F)];
}

// dec_in_tm[t][b][c] = (t==0) ? 0 : labels[b][t][c]
__global__ void build_dec_in_k(const float* __restrict__ labels, float* __restrict__ out) {
  long long total = (long long)SC_LEN * B_SZ * N_CHARS_;
  long long idx = (long long)blockIdx.x * blockDim.x + threadIdx.x;
  if (idx >= total) return;
  int c = (int)(idx % N_CHARS_);
  long long sb = idx / N_CHARS_;
  int b = (int)(sb % B_SZ);
  int s = (int)(sb / B_SZ);
  out[idx] = (s == 0) ? 0.f : labels[((long long)b * SC_LEN + s) * N_CHARS_ + c];
}

// Torch LSTM gate order i,f,g,o. gates = h@Whh^T, xpre = x@Wih^T + b.
__global__ void lstm_step_k(const float* __restrict__ gates, const float* __restrict__ xpre,
                            float* __restrict__ h, float* __restrict__ c,
                            float* __restrict__ out, int H, int ldh, int ldo) {
  int idx = blockIdx.x * blockDim.x + threadIdx.x;
  if (idx >= B_SZ * H) return;
  int b = idx / H, j = idx % H;
  long long g0 = (long long)b * 4 * H;
  float gi = gates[g0 + j]         + xpre[g0 + j];
  float gf = gates[g0 + H + j]     + xpre[g0 + H + j];
  float gg = gates[g0 + 2 * H + j] + xpre[g0 + 2 * H + j];
  float go = gates[g0 + 3 * H + j] + xpre[g0 + 3 * H + j];
  float i_ = 1.f / (1.f + __expf(-gi));
  float f_ = 1.f / (1.f + __expf(-gf));
  float gt = tanhf(gg);
  float o_ = 1.f / (1.f + __expf(-go));
  long long hio = (long long)b * ldh + j;
  float cc = f_ * c[hio] + i_ * gt;
  float hh = o_ * tanhf(cc);
  c[hio] = cc;
  h[hio] = hh;
  out[(long long)b * ldo + j] = hh;
}

__global__ void softmax_rows_k(float* __restrict__ x, int rows, int L) {
  int r = blockIdx.x * blockDim.x + threadIdx.x;
  if (r >= rows) return;
  float* p = x + (long long)r * L;
  float mx = -3.4e38f;
  for (int i = 0; i < L; ++i) mx = fmaxf(mx, p[i]);
  float s = 0.f;
  for (int i = 0; i < L; ++i) { float e = __expf(p[i] - mx); p[i] = e; s += e; }
  float inv = 1.f / s;
  for (int i = 0; i < L; ++i) p[i] *= inv;
}

// Exact 1.5-entmax over 128 logits; input [SC,B,128], output [B,SC,128].
__global__ __launch_bounds__(128) void entmax15_k(const float* __restrict__ logits,
                                                  float* __restrict__ out) {
  __shared__ float xs[128];
  __shared__ float srt[128];
  __shared__ float red[128];
  __shared__ float tau[128];
  __shared__ float taustar;
  int row = blockIdx.x;                 // row = s*B + b
  int s = row / B_SZ, b = row % B_SZ;
  int t = threadIdx.x;
  float v = logits[(long long)row * N_CHARS_ + t] * 0.5f;
  red[t] = v; __syncthreads();
  for (int o = 64; o > 0; o >>= 1) {
    if (t < o) red[t] = fmaxf(red[t], red[t + o]);
    __syncthreads();
  }
  v -= red[0];
  xs[t] = v; srt[t] = v;
  __syncthreads();
  // bitonic sort, descending
  for (int k = 2; k <= 128; k <<= 1) {
    for (int j = k >> 1; j > 0; j >>= 1) {
      int ixj = t ^ j;
      if (ixj > t) {
        float a = srt[t], c = srt[ixj];
        bool descend = ((t & k) == 0);
        bool sw = descend ? (a < c) : (a > c);
        if (sw) { srt[t] = c; srt[ixj] = a; }
      }
      __syncthreads();
    }
  }
  if (t == 0) {
    float csum = 0.f, css = 0.f; int cnt = 0;
    for (int k2 = 0; k2 < 128; ++k2) {
      float xk = srt[k2];
      csum += xk; css += xk * xk;
      float kk = (float)(k2 + 1);
      float mean = csum / kk;
      float ss = css - csum * mean;
      float val = fmaxf((1.f - ss) / kk, 0.f);
      float tk = mean - sqrtf(val);
      tau[k2] = tk;
      if (tk <= xk) cnt++;
    }
    if (cnt < 1) cnt = 1;
    taustar = tau[cnt - 1];
  }
  __syncthreads();
  float d = fmaxf(xs[t] - taustar, 0.f);
  out[((long long)b * SC_LEN + s) * N_CHARS_ + t] = d * d;
}

static inline void gemm(bool transB, const float* A, const float* Bm, const float* bias,
                        float* C, int M, int N, int K, int lda, int ldb, int ldc,
                        long long sA, long long sB, long long sC, int batch, float alpha,
                        hipStream_t st) {
  dim3 g((N + BN - 1) / BN, (M + BM - 1) / BM, batch);
  if (transB)
    gemm_bf16_wmma<true><<<g, 256, 0, st>>>(A, Bm, bias, C, M, N, K, lda, ldb, ldc, sA, sB, sC, alpha);
  else
    gemm_bf16_wmma<false><<<g, 256, 0, st>>>(A, Bm, bias, C, M, N, K, lda, ldb, ldc, sA, sB, sC, alpha);
}

#define CDIV(a, b) (((a) + (b) - 1) / (b))

extern "C" void kernel_launch(void* const* d_in, const int* in_sizes, int n_in,
                              void* d_out, int out_size, void* d_ws, size_t ws_size,
                              hipStream_t stream) {
  (void)in_sizes; (void)n_in; (void)out_size; (void)ws_size;
  const int*   li        = (const int*)d_in[0];
  const float* char_seq  = (const float*)d_in[1];
  const float* tagset    = (const float*)d_in[2];
  const float* labels    = (const float*)d_in[3];
  const float* lang_emb  = (const float*)d_in[4];
  const float* c_wih_f   = (const float*)d_in[5];
  const float* c_whh_f   = (const float*)d_in[6];
  const float* c_b_f     = (const float*)d_in[7];
  const float* c_wih_b   = (const float*)d_in[8];
  const float* c_whh_b   = (const float*)d_in[9];
  const float* c_b_b     = (const float*)d_in[10];
  const float* t_wih_f   = (const float*)d_in[11];
  const float* t_whh_f   = (const float*)d_in[12];
  const float* t_b_f     = (const float*)d_in[13];
  const float* t_wih_b   = (const float*)d_in[14];
  const float* t_whh_b   = (const float*)d_in[15];
  const float* t_b_b     = (const float*)d_in[16];
  const float* cell_wih  = (const float*)d_in[17];
  const float* cell_whh  = (const float*)d_in[18];
  const float* cell_b    = (const float*)d_in[19];
  const float* c_inw     = (const float*)d_in[20];
  const float* c_inb     = (const float*)d_in[21];
  const float* c_outw    = (const float*)d_in[22];
  const float* c_outb    = (const float*)d_in[23];
  const float* t_inw     = (const float*)d_in[24];
  const float* t_inb     = (const float*)d_in[25];
  const float* t_outw    = (const float*)d_in[26];
  const float* t_outb    = (const float*)d_in[27];
  const float* fc_w      = (const float*)d_in[28];
  const float* fc_b      = (const float*)d_in[29];

  char* ws = (char*)d_ws;
  size_t off = 0;
  auto alloc = [&](size_t nfloats) -> float* {
    float* p = (float*)(ws + off);
    off += ((nfloats * sizeof(float)) + 255) & ~(size_t)255;
    return p;
  };

  float* emb       = alloc((size_t)B_SZ * LANG_DIM_);
  float* cs_tm     = alloc((size_t)SC_LEN * B_SZ * 192);        // char input, time-major
  float* ts_tm     = alloc((size_t)ST_LEN * B_SZ * 128);        // tag input, time-major
  float* dec_in    = alloc((size_t)SC_LEN * B_SZ * N_CHARS_);   // shifted labels, time-major
  float* cxf       = alloc((size_t)SC_LEN * B_SZ * 1024);       // char fwd x@Wih^T + b
  float* cxb       = alloc((size_t)SC_LEN * B_SZ * 1024);
  float* txf       = alloc((size_t)ST_LEN * B_SZ * 1024);
  float* txb       = alloc((size_t)ST_LEN * B_SZ * 1024);
  float* dxp       = alloc((size_t)SC_LEN * B_SZ * 2048);       // decoder x@Wih^T + b
  float* char_out  = alloc((size_t)SC_LEN * B_SZ * E_DIM);      // [S,B,512] fwd|bwd halves
  float* tag_out   = alloc((size_t)ST_LEN * B_SZ * E_DIM);
  float* dec_hs    = alloc((size_t)SC_LEN * B_SZ * E_DIM);
  float* dec_h     = alloc((size_t)B_SZ * E_DIM);               // also hosts encoder fwd finals
  float* dec_c     = alloc((size_t)B_SZ * E_DIM);
  float* hb_c      = alloc((size_t)B_SZ * H_DIM);
  float* cb_c      = alloc((size_t)B_SZ * H_DIM);
  float* hb_t      = alloc((size_t)B_SZ * H_DIM);
  float* cb_t      = alloc((size_t)B_SZ * H_DIM);
  float* gates     = alloc((size_t)B_SZ * 2048);
  float* qh        = alloc((size_t)SC_LEN * B_SZ * E_DIM);
  float* kh        = alloc((size_t)SC_LEN * B_SZ * E_DIM);
  float* vh        = alloc((size_t)SC_LEN * B_SZ * E_DIM);
  float* scores    = alloc((size_t)B_SZ * SC_LEN * SC_LEN);
  float* apre      = alloc((size_t)SC_LEN * B_SZ * E_DIM);
  float* cat_buf   = alloc((size_t)SC_LEN * B_SZ * 1024);
  float* logits    = alloc((size_t)SC_LEN * B_SZ * N_CHARS_);

  const long long BE = (long long)B_SZ * E_DIM;     // 262144
  const float inv_sqrt_e = 0.044194173824159216f;   // 1/sqrt(512)

  // 1) embeddings + input assembly
  build_emb_k<<<CDIV(B_SZ * LANG_DIM_, 256), 256, 0, stream>>>(li, lang_emb, emb);
  {
    long long tc = (long long)SC_LEN * B_SZ * 192;
    concat_tm_k<<<(unsigned)CDIV(tc, 256), 256, 0, stream>>>(char_seq, emb, cs_tm, SC_LEN, N_CHARS_);
    long long tt = (long long)ST_LEN * B_SZ * 128;
    concat_tm_k<<<(unsigned)CDIV(tt, 256), 256, 0, stream>>>(tagset, emb, ts_tm, ST_LEN, N_TAGS_);
    long long td = (long long)SC_LEN * B_SZ * N_CHARS_;
    build_dec_in_k<<<(unsigned)CDIV(td, 256), 256, 0, stream>>>(labels, dec_in);
  }

  // 2) input projections (big batched WMMA GEMMs), bias folded in
  gemm(true, cs_tm, c_wih_f, c_b_f, cxf, SC_LEN * B_SZ, 1024, 192, 192, 192, 1024, 0, 0, 0, 1, 1.f, stream);
  gemm(true, cs_tm, c_wih_b, c_b_b, cxb, SC_LEN * B_SZ, 1024, 192, 192, 192, 1024, 0, 0, 0, 1, 1.f, stream);
  gemm(true, ts_tm, t_wih_f, t_b_f, txf, ST_LEN * B_SZ, 1024, 128, 128, 128, 1024, 0, 0, 0, 1, 1.f, stream);
  gemm(true, ts_tm, t_wih_b, t_b_b, txb, ST_LEN * B_SZ, 1024, 128, 128, 128, 1024, 0, 0, 0, 1, 1.f, stream);
  gemm(true, dec_in, cell_wih, cell_b, dxp, SC_LEN * B_SZ, 2048, 128, 128, 128, 2048, 0, 0, 0, 1, 1.f, stream);

  // 3) zero initial states (encoder fwd states live inside dec_h/dec_c halves)
  hipMemsetAsync(dec_h, 0, (size_t)B_SZ * E_DIM * 4, stream);
  hipMemsetAsync(dec_c, 0, (size_t)B_SZ * E_DIM * 4, stream);
  hipMemsetAsync(hb_c, 0, (size_t)B_SZ * H_DIM * 4, stream);
  hipMemsetAsync(cb_c, 0, (size_t)B_SZ * H_DIM * 4, stream);
  hipMemsetAsync(hb_t, 0, (size_t)B_SZ * H_DIM * 4, stream);
  hipMemsetAsync(cb_t, 0, (size_t)B_SZ * H_DIM * 4, stream);

  const int stepsC = CDIV(B_SZ * H_DIM, 256);
  const int stepsD = CDIV(B_SZ * E_DIM, 256);

  // 4) char BiLSTM recurrences (forward h/c = columns 0..255 of dec_h/dec_c)
  for (int t = 0; t < SC_LEN; ++t) {
    gemm(true, dec_h, c_whh_f, nullptr, gates, B_SZ, 1024, H_DIM, E_DIM, H_DIM, 1024, 0, 0, 0, 1, 1.f, stream);
    lstm_step_k<<<stepsC, 256, 0, stream>>>(gates, cxf + (long long)t * B_SZ * 1024,
        dec_h, dec_c, char_out + (long long)t * B_SZ * E_DIM, H_DIM, E_DIM, E_DIM);
  }
  for (int t = 0; t < SC_LEN; ++t) {
    int s = SC_LEN - 1 - t;
    gemm(true, hb_c, c_whh_b, nullptr, gates, B_SZ, 1024, H_DIM, H_DIM, H_DIM, 1024, 0, 0, 0, 1, 1.f, stream);
    lstm_step_k<<<stepsC, 256, 0, stream>>>(gates, cxb + (long long)s * B_SZ * 1024,
        hb_c, cb_c, char_out + (long long)s * B_SZ * E_DIM + H_DIM, H_DIM, H_DIM, E_DIM);
  }

  // 5) tag BiLSTM (forward h/c = columns 256..511 of dec_h/dec_c)
  for (int t = 0; t < ST_LEN; ++t) {
    gemm(true, dec_h + H_DIM, t_whh_f, nullptr, gates, B_SZ, 1024, H_DIM, E_DIM, H_DIM, 1024, 0, 0, 0, 1, 1.f, stream);
    lstm_step_k<<<stepsC, 256, 0, stream>>>(gates, txf + (long long)t * B_SZ * 1024,
        dec_h + H_DIM, dec_c + H_DIM, tag_out + (long long)t * B_SZ * E_DIM, H_DIM, E_DIM, E_DIM);
  }
  for (int t = 0; t < ST_LEN; ++t) {
    int s = ST_LEN - 1 - t;
    gemm(true, hb_t, t_whh_b, nullptr, gates, B_SZ, 1024, H_DIM, H_DIM, H_DIM, 1024, 0, 0, 0, 1, 1.f, stream);
    lstm_step_k<<<stepsC, 256, 0, stream>>>(gates, txb + (long long)s * B_SZ * 1024,
        hb_t, cb_t, tag_out + (long long)s * B_SZ * E_DIM + H_DIM, H_DIM, H_DIM, E_DIM);
  }

  // 6) decoder LSTMCell (h0/c0 already in dec_h/dec_c from encoder finals)
  for (int t = 0; t < SC_LEN; ++t) {
    gemm(true, dec_h, cell_whh, nullptr, gates, B_SZ, 2048, E_DIM, E_DIM, E_DIM, 2048, 0, 0, 0, 1, 1.f, stream);
    lstm_step_k<<<stepsD, 256, 0, stream>>>(gates, dxp + (long long)t * B_SZ * 2048,
        dec_h, dec_c, dec_hs + (long long)t * B_SZ * E_DIM, E_DIM, E_DIM, E_DIM);
  }

  // 7) attention (char then tag, reusing qh/kh/vh/scores/apre)
  const int MQ = SC_LEN * B_SZ;   // 24576 query rows
  for (int a = 0; a < 2; ++a) {
    const bool is_char = (a == 0);
    const float* inw  = is_char ? c_inw  : t_inw;
    const float* inb  = is_char ? c_inb  : t_inb;
    const float* outw = is_char ? c_outw : t_outw;
    const float* outb = is_char ? c_outb : t_outb;
    const float* kv   = is_char ? char_out : tag_out;
    const int Lk      = is_char ? SC_LEN : ST_LEN;
    const int MK      = Lk * B_SZ;
    float* cdst       = cat_buf + (is_char ? 0 : E_DIM);

    gemm(true, dec_hs, inw,                inb,          qh, MQ, E_DIM, E_DIM, E_DIM, E_DIM, E_DIM, 0, 0, 0, 1, 1.f, stream);
    gemm(true, kv,     inw + 512 * 512,    inb + 512,    kh, MK, E_DIM, E_DIM, E_DIM, E_DIM, E_DIM, 0, 0, 0, 1, 1.f, stream);
    gemm(true, kv,     inw + 1024 * 512,   inb + 1024,   vh, MK, E_DIM, E_DIM, E_DIM, E_DIM, E_DIM, 0, 0, 0, 1, 1.f, stream);
    // scores[b,q,k] = qh[q,b,:] . kh[k,b,:] / sqrt(E)   (batched over b)
    gemm(true, qh, kh, nullptr, scores, SC_LEN, Lk, E_DIM,
         (int)BE, (int)BE, Lk, E_DIM, E_DIM, (long long)SC_LEN * Lk, B_SZ, inv_sqrt_e, stream);
    softmax_rows_k<<<CDIV(B_SZ * SC_LEN, 256), 256, 0, stream>>>(scores, B_SZ * SC_LEN, Lk);
    // apre[q,b,:] = sum_k attn[b,q,k] * vh[k,b,:]
    gemm(false, scores, vh, nullptr, apre, SC_LEN, E_DIM, Lk,
         Lk, (int)BE, (int)BE, (long long)SC_LEN * Lk, E_DIM, E_DIM, B_SZ, 1.f, stream);
    // out-projection written straight into the concat buffer half
    gemm(true, apre, outw, outb, cdst, MQ, E_DIM, E_DIM, E_DIM, E_DIM, 1024, 0, 0, 0, 1, 1.f, stream);
  }

  // 8) FC + entmax1.5 (output transposed to [B,SC,128])
  gemm(true, cat_buf, fc_w, fc_b, logits, MQ, N_CHARS_, 1024, 1024, 1024, N_CHARS_, 0, 0, 0, 1, 1.f, stream);
  entmax15_k<<<SC_LEN * B_SZ, 128, 0, stream>>>(logits, (float*)d_out);
}
